// ParsevalAttention_73031623901189
// MI455X (gfx1250) — matrix operand order repaired and checked by reading, not compiled, not run on hardware
//
#include <hip/hip_runtime.h>
#include <math.h>

// ---------------- problem constants ----------------
#define B_BATCH 4
#define T_SEQ   2048      // 2^11
#define C_DIM   256
#define HALF_C  128
#define NEARW   16
#define SCALE   (1.0f/16.0f)   // 1/sqrt(C)

typedef __attribute__((ext_vector_type(16))) __bf16 v16bf;
typedef __attribute__((ext_vector_type(8)))  __bf16 v8bf;
typedef __attribute__((ext_vector_type(8)))  float  v8f;

// ================= WMMA fragment helpers (CDNA5 wave32 layouts) =============
// A (16x32 bf16, row-major source [M][K]): lane L -> M=L&15,
//   elems 0..7 = K kbase+khalf+0..7, elems 8..15 = K kbase+16+khalf+0..7.
__device__ __forceinline__ v16bf load_a_frag(const __bf16* __restrict__ A,
                                             int lda, int kbase) {
  int lane  = threadIdx.x & 31;
  int m     = lane & 15;
  int khalf = (lane >> 4) << 3;                   // 0 or 8
  const __bf16* p = A + (size_t)m * lda + kbase + khalf;
  v8bf lo = *(const v8bf*)p;
  v8bf hi = *(const v8bf*)(p + 16);
  v16bf r;
#pragma unroll
  for (int i = 0; i < 8; ++i) { r[i] = lo[i]; r[i + 8] = hi[i]; }
  return r;
}

// B (32x16 bf16, source packed N-major [N][K]): lane L -> N=L&15,
//   elems 0..7 = K kbase+khalf+0..7, elems 8..15 = K kbase+16+khalf+0..7.
__device__ __forceinline__ v16bf load_b_frag(const __bf16* Bm, int ldb,
                                             int kbase) {
  int lane = threadIdx.x & 31;
  int n    = lane & 15;
  int koff = (lane >> 4) << 3;                    // 0 or 8
  const __bf16* p = Bm + (size_t)n * ldb + kbase + koff;
  v8bf lo = *(const v8bf*)p;
  v8bf hi = *(const v8bf*)(p + 16);
  v16bf r;
#pragma unroll
  for (int i = 0; i < 8; ++i) { r[i] = lo[i]; r[i + 8] = hi[i]; }
  return r;
}

__device__ __forceinline__ v8f wmma_bf16(v16bf a, v16bf b, v8f c) {
  return __builtin_amdgcn_wmma_f32_16x16x32_bf16(false, a, false, b,
                                                 (short)0, c, false, false);
}

// C/D f32 16x16: lane L -> N=L&15, VGPR r -> M = r + 8*(L>=16).
__device__ __forceinline__ void store_tile_f32(float* __restrict__ Cp, int ldc,
                                               const v8f acc) {
  int lane = threadIdx.x & 31;
  int n    = lane & 15;
  int moff = (lane >> 4) << 3;
#pragma unroll
  for (int r = 0; r < 8; ++r) Cp[(size_t)(moff + r) * ldc + n] = acc[r];
}

// ================= Tensor Data Mover: 2D bf16 tile -> LDS offset 0 ==========
#if __has_builtin(__builtin_amdgcn_tensor_load_to_lds)
#define HAVE_TDM 1
typedef __attribute__((ext_vector_type(4))) unsigned int u32x4;
typedef __attribute__((ext_vector_type(8))) int i32x8;
typedef __attribute__((ext_vector_type(4))) int i32x4;

// Load `rows` rows of `row_elems` bf16 (row stride `stride_elems`) from
// `gaddr` into LDS address 0.  D# per CDNA5 ISA ch.8 (group0 + group1, 2D).
__device__ __forceinline__ void tdm_load_2d_lds0(const void* gaddr, int rows,
                                                 int row_elems,
                                                 int stride_elems) {
  unsigned long long ga = (unsigned long long)gaddr;
  u32x4 g0;
  g0[0] = 1u;                                   // count=1, user descriptor
  g0[1] = 0u;                                   // lds_addr = 0
  g0[2] = (unsigned int)(ga & 0xFFFFFFFFull);   // global_addr[31:0]
  g0[3] = (unsigned int)((ga >> 32) & 0x01FFFFFFull) | (2u << 30); // +type=2
  i32x8 g1;
  g1[0] = (int)(1u << 16);                      // data_size=1 (2 bytes)
  g1[1] = (int)((row_elems & 0xFFFF) << 16);    // tensor_dim0[15:0]
  g1[2] = (int)(((unsigned)row_elems >> 16) |   // tensor_dim0[31:16]
                ((unsigned)(rows & 0xFFFF) << 16));   // tensor_dim1[15:0]
  g1[3] = (int)(((unsigned)rows >> 16) |        // tensor_dim1[31:16]
                ((unsigned)(row_elems & 0xFFFF) << 16)); // tile_dim0
  g1[4] = (int)(rows & 0xFFFF);                 // tile_dim1 (tile_dim2=0)
  g1[5] = stride_elems;                         // tensor_dim0_stride[31:0]
  g1[6] = 0;                                    // stride hi + dim1_stride lo
  g1[7] = 0;
  i32x4 z4 = {0, 0, 0, 0};
  i32x8 z8 = {0, 0, 0, 0, 0, 0, 0, 0};
  // 6-arg form (this toolchain): (g0, g1, g2, g3, g4, cpol)
  __builtin_amdgcn_tensor_load_to_lds(g0, g1, z4, z4, z8, 0);
}
#endif

// stage a 64x256 bf16 N-major tile (32KB) into LDS (Btile must sit at LDS 0)
__device__ __forceinline__ void stage_b_tile(__bf16* Btile,
                                             const __bf16* __restrict__ src) {
#ifdef HAVE_TDM
  if ((threadIdx.x >> 5) == 0) {
    tdm_load_2d_lds0(src, 64, 256, 256);
    __builtin_amdgcn_s_wait_tensorcnt(0);
  }
  // The TDM writes LDS behind the compiler's back.  Escape the pointer into
  // an asm memory clobber so loads from Btile are treated as reading
  // externally-initialized memory (prevents poison-folding / bogus hoisting).
  asm volatile("" : : "r"(Btile) : "memory");
#else
  for (int i = threadIdx.x; i < 64 * 256 / 8; i += 256)
    ((v8bf*)Btile)[i] = ((const v8bf*)src)[i];
#endif
  __syncthreads();
}

// ======================= K0: convert x -> bf16, rope tables ==================
__global__ void __launch_bounds__(256) prep_kernel(const float* __restrict__ x,
                                                   __bf16* __restrict__ xbf,
                                                   float* __restrict__ cosT,
                                                   float* __restrict__ sinT) {
  int idx    = blockIdx.x * blockDim.x + threadIdx.x;
  int stride = gridDim.x * blockDim.x;
  const int total = B_BATCH * T_SEQ * C_DIM;
  for (int i = idx; i < total; i += stride) xbf[i] = (__bf16)x[i];
  const int nrt = T_SEQ * HALF_C;
  for (int i = idx; i < nrt; i += stride) {
    int t = i >> 7, h = i & (HALF_C - 1);
    float inv_freq = powf(10000.0f, -(float)h * (1.0f / (float)HALF_C));
    float ang = (float)t * inv_freq;
    cosT[i] = cosf(ang);
    sinT[i] = sinf(ang);
  }
}

// ======================= K1: Gauss-Jordan inverse (256x256) ==================
__global__ void __launch_bounds__(256) gj_inverse_kernel(
    const float* __restrict__ W, float* __restrict__ A) {
  const int N = 256;
  int j = threadIdx.x;
  for (int i = 0; i < N; ++i) A[i * N + j] = W[i * N + j];
  __syncthreads();
  __shared__ float prow[256];
  __shared__ float pcol[256];
  __shared__ float pivinv_s;
  for (int p = 0; p < N; ++p) {
    pcol[j] = A[j * N + p];
    if (j == 0) pivinv_s = 1.0f / A[p * N + p];
    __syncthreads();
    float pivinv = pivinv_s;
    float rp = (j == p) ? pivinv : A[p * N + j] * pivinv;
    prow[j] = rp;
    __syncthreads();
    for (int i = 0; i < N; ++i) {
      if (i == p) { A[p * N + j] = rp; continue; }
      float f = pcol[i];
      A[i * N + j] = (j == p) ? (-f * pivinv)
                              : fmaf(-f, prow[j], A[i * N + j]);
    }
    __syncthreads();
  }
}

// ======================= K2: pack weights to bf16 N-major ====================
__global__ void __launch_bounds__(256) pack_weights_kernel(
    const float* __restrict__ W_Q, const float* __restrict__ Ainv,
    const float* __restrict__ W_V, const float* __restrict__ W_O,
    __bf16* __restrict__ Wq, __bf16* __restrict__ Wk,
    __bf16* __restrict__ Wv, __bf16* __restrict__ Wo) {
  int idx = blockIdx.x * 256 + threadIdx.x;   // [n][k], 65536 total
  int n = idx >> 8, k = idx & 255;
  Wq[idx] = (__bf16)W_Q[k * 256 + n];
  Wk[idx] = (__bf16)Ainv[idx];
  Wv[idx] = (__bf16)W_V[idx];
  Wo[idx] = (__bf16)W_O[idx];
}

// ======================= K3: QKV projection GEMM (TDM + WMMA) ================
// block: 8 waves, M-block 128, N-block 64 (4 sub-tiles / wave, A reused)
__global__ void __launch_bounds__(256) gemm_qkv_kernel(
    const __bf16* __restrict__ xbf, const __bf16* __restrict__ Wq,
    const __bf16* __restrict__ Wk, const __bf16* __restrict__ Wv,
    float* __restrict__ qraw, float* __restrict__ kraw,
    __bf16* __restrict__ vT) {
  __shared__ __bf16 Btile[64 * 256];            // 32KB, LDS offset 0
  int wave = threadIdx.x >> 5;
  int n0 = blockIdx.x * 64;
  int m0 = blockIdx.y * 128 + wave * 16;        // global row in [0, B*T)
  int which = blockIdx.z;                       // 0=q 1=k 2=v
  const __bf16* Bmat = (which == 0) ? Wq : (which == 1) ? Wk : Wv;
  stage_b_tile(Btile, Bmat + (size_t)n0 * C_DIM);

  const __bf16* Ap = xbf + (size_t)m0 * C_DIM;
  v8f acc[4] = {};
#pragma unroll
  for (int k0 = 0; k0 < C_DIM; k0 += 32) {
    v16bf a = load_a_frag(Ap, C_DIM, k0);
#pragma unroll
    for (int s = 0; s < 4; ++s) {
      v16bf b = load_b_frag(Btile + s * 16 * 256, 256, k0);
      acc[s] = wmma_bf16(a, b, acc[s]);
    }
  }
  int lane = threadIdx.x & 31;
  int nn = lane & 15, moff = (lane >> 4) << 3;
  if (which < 2) {
    float* base = ((which == 0) ? qraw : kraw) + (size_t)m0 * C_DIM + n0;
#pragma unroll
    for (int s = 0; s < 4; ++s) store_tile_f32(base + s * 16, C_DIM, acc[s]);
  } else {
#pragma unroll
    for (int s = 0; s < 4; ++s)
#pragma unroll
      for (int r = 0; r < 8; ++r) {
        int M = m0 + moff + r;
        int b = M >> 11, t = M & (T_SEQ - 1);
        vT[(size_t)b * C_DIM * T_SEQ + (size_t)(n0 + s * 16 + nn) * T_SEQ + t] =
            (__bf16)acc[s][r];
      }
  }
}

// ======================= K4: RoPE + L2 normalize -> bf16 =====================
__global__ void __launch_bounds__(256) rope_norm_kernel(
    const float* __restrict__ qraw, const float* __restrict__ kraw,
    const float* __restrict__ cosT, const float* __restrict__ sinT,
    __bf16* __restrict__ qrow, __bf16* __restrict__ krow) {
  int tok = blockIdx.x;                 // 0 .. B*T-1
  int c = threadIdx.x;
  int t = tok & (T_SEQ - 1);
  int h = c & (HALF_C - 1);
  float cs = cosT[t * HALF_C + h], sn = sinT[t * HALF_C + h];
  __shared__ float red[256];
  __shared__ float scale_s;
  {
    const float* p = qraw + (size_t)tok * C_DIM;
    float x1 = p[h], x2 = p[h + HALF_C];
    float val = (c < HALF_C) ? (x1 * cs - x2 * sn) : (x1 * sn + x2 * cs);
    red[c] = val * val;
    __syncthreads();
    for (int off = 128; off > 0; off >>= 1) {
      if (c < off) red[c] += red[c + off];
      __syncthreads();
    }
    if (c == 0) scale_s = 1.0f / (sqrtf(red[0]) + 1e-8f);
    __syncthreads();
    qrow[(size_t)tok * C_DIM + c] = (__bf16)(val * scale_s);
    __syncthreads();
  }
  {
    const float* p = kraw + (size_t)tok * C_DIM;
    float x1 = p[h], x2 = p[h + HALF_C];
    float val = (c < HALF_C) ? (x1 * cs - x2 * sn) : (x1 * sn + x2 * cs);
    red[c] = val * val;
    __syncthreads();
    for (int off = 128; off > 0; off >>= 1) {
      if (c < off) red[c] += red[c + off];
      __syncthreads();
    }
    if (c == 0) scale_s = 1.0f / (sqrtf(red[0]) + 1e-8f);
    __syncthreads();
    krow[(size_t)tok * C_DIM + c] = (__bf16)(val * scale_s);
  }
}

// ======================= K5: Haar coefficients (7 cols, sparse) ==============
__global__ void __launch_bounds__(256) haar_coef_kernel(
    const __bf16* __restrict__ qrow, const __bf16* __restrict__ krow,
    float* __restrict__ qcoef, float* __restrict__ kcoef) {
  int b = blockIdx.y;
  int wcol = blockIdx.x;                // 0..6
  int c = threadIdx.x;
  int lev = (wcol == 0) ? 0 : (wcol < 3) ? 1 : 2;
  int kidx = wcol - ((1 << lev) - 1);
  int bs = T_SEQ >> lev, half = bs >> 1, s0 = kidx * bs;
  float inv = rsqrtf((float)half);
  float aq = 0.f, ak = 0.f;
  for (int t = s0; t < s0 + half; ++t) {
    size_t o = ((size_t)b * T_SEQ + t) * C_DIM + c;
    aq += (float)qrow[o];
    ak += (float)krow[o];
  }
  for (int t = s0 + half; t < s0 + bs; ++t) {
    size_t o = ((size_t)b * T_SEQ + t) * C_DIM + c;
    aq -= (float)qrow[o];
    ak -= (float)krow[o];
  }
  qcoef[(b * 7 + wcol) * C_DIM + c] = aq * inv;
  kcoef[(b * 7 + wcol) * C_DIM + c] = ak * inv;
}

// ======================= K6: attn_coef (B x 7 x 7) ===========================
__global__ void __launch_bounds__(256) coef_gemm_kernel(
    const float* __restrict__ qcoef, const float* __restrict__ kcoef,
    float* __restrict__ coef) {
  int b = blockIdx.x;
  int tid = threadIdx.x;
  __shared__ float qs[7][256], ks[7][256];
  for (int w = 0; w < 7; ++w) {
    qs[w][tid] = qcoef[(b * 7 + w) * C_DIM + tid];
    ks[w][tid] = kcoef[(b * 7 + w) * C_DIM + tid];
  }
  __syncthreads();
  if (tid < 49) {
    int w1 = tid / 7, w2 = tid % 7;
    float s = 0.f;
    for (int c = 0; c < C_DIM; ++c) s += qs[w1][c] * ks[w2][c];
    coef[b * 49 + tid] = s * SCALE;
  }
}

// ======================= K7a: scores + variance softmax -> P bf16 ============
__global__ void __launch_bounds__(256) score_softmax_kernel(
    const __bf16* __restrict__ qrow, const __bf16* __restrict__ krow,
    const float* __restrict__ coef, __bf16* __restrict__ P) {
  int b = blockIdx.y;
  int t0 = blockIdx.x * 16;
  int tid = threadIdx.x;
  __shared__ float cf[49];
  __shared__ float nearS[16][17];
  __shared__ float at[16][8];
  __shared__ float redA[16][16], redB[16][16], redC[16][16];

  if (tid < 49) cf[tid] = coef[b * 49 + tid];
  __syncthreads();

  for (int task = tid; task < 16 * 17; task += 256) {
    int row = task / 17, d = task % 17;
    int t = t0 + row, s = t - d;
    float acc = 0.f;
    if (s >= 0) {
      const __bf16* qp = qrow + ((size_t)b * T_SEQ + t) * C_DIM;
      const __bf16* kp = krow + ((size_t)b * T_SEQ + s) * C_DIM;
      for (int c = 0; c < C_DIM; ++c) acc += (float)qp[c] * (float)kp[c];
    }
    nearS[row][d] = acc * SCALE;
  }
  if (tid < 16) {
    int tt = t0 + tid;
    float av[7] = {0, 0, 0, 0, 0, 0, 0};
#pragma unroll
    for (int j = 0; j < 3; ++j) {
      int bs = T_SEQ >> j, half = bs >> 1;
      int w1 = ((1 << j) - 1) + (tt >> (11 - j));
      float inv = rsqrtf((float)half);
      float hv = ((tt & (bs - 1)) < half) ? inv : -inv;
      for (int w2 = 0; w2 < 7; ++w2) av[w2] += hv * cf[w1 * 7 + w2];
    }
    for (int w2 = 0; w2 < 7; ++w2) at[tid][w2] = av[w2];
  }
  __syncthreads();

  int row = tid >> 4, l = tid & 15;
  int t = t0 + row, n = t + 1;
  float rnv = rsqrtf((float)n);

  auto val_at = [&](int s) -> float {
    int d = t - s;
    float raw;
    if (d <= NEARW) {
      raw = nearS[row][d];
    } else {
      raw = 0.f;
#pragma unroll
      for (int j = 0; j < 3; ++j) {
        int bs = T_SEQ >> j, half = bs >> 1;
        int w = ((1 << j) - 1) + (s >> (11 - j));
        float inv = rsqrtf((float)half);
        float hv = ((s & (bs - 1)) < half) ? inv : -inv;
        raw += hv * at[row][w];
      }
    }
    return raw * rnv;
  };

  float sum = 0.f, sumsq = 0.f, vmax = -1e30f;
  for (int s = l; s <= t; s += 16) {
    float v = val_at(s);
    sum += v; sumsq += v * v; vmax = fmaxf(vmax, v);
  }
  redA[row][l] = sum; redB[row][l] = sumsq; redC[row][l] = vmax;
  __syncthreads();
  for (int off = 8; off > 0; off >>= 1) {
    if (l < off) {
      redA[row][l] += redA[row][l + off];
      redB[row][l] += redB[row][l + off];
      redC[row][l] = fmaxf(redC[row][l], redC[row][l + off]);
    }
    __syncthreads();
  }
  float mean = redA[row][0] / (float)n;
  float var = redB[row][0] / (float)n - mean * mean;
  float istd = rsqrtf(fmaxf(var, 1e-6f));
  float zmax = (redC[row][0] - mean) * istd;
  __syncthreads();

  float esum = 0.f;
  for (int s = l; s <= t; s += 16)
    esum += __expf((val_at(s) - mean) * istd - zmax);
  redA[row][l] = esum;
  __syncthreads();
  for (int off = 8; off > 0; off >>= 1) {
    if (l < off) redA[row][l] += redA[row][l + off];
    __syncthreads();
  }
  float inv_esum = 1.0f / redA[row][0];

  __bf16* Pp = P + ((size_t)b * T_SEQ + t) * T_SEQ;
  int cover = (t0 + 47) & ~31;
  for (int s = l; s < cover; s += 16) {
    float pv = 0.f;
    if (s <= t) pv = __expf((val_at(s) - mean) * istd - zmax) * inv_esum;
    Pp[s] = (__bf16)pv;
  }
}

// ======================= K7b: y = P @ v (triangular WMMA, 16x64/wave) ========
__global__ void __launch_bounds__(256) attn_v_kernel(
    const __bf16* __restrict__ P, const __bf16* __restrict__ vT,
    __bf16* __restrict__ y) {
  int b = blockIdx.z;
  int wave = threadIdx.x >> 5;
  int m0 = blockIdx.y * 128 + wave * 16;   // t tile
  int n0 = blockIdx.x * 64;                // c tile (4 sub-tiles)
  const __bf16* Ap = P + (size_t)b * T_SEQ * T_SEQ + (size_t)m0 * T_SEQ;
  const __bf16* Bp = vT + (size_t)b * C_DIM * T_SEQ + (size_t)n0 * T_SEQ;
  int kend = (m0 + 47) & ~31;              // causal cover
  v8f acc[4] = {};
  for (int k0 = 0; k0 < kend; k0 += 32) {
    __builtin_prefetch(Ap + (size_t)(threadIdx.x & 15) * T_SEQ + k0 + 64, 0, 3);
    v16bf a = load_a_frag(Ap, T_SEQ, k0);
#pragma unroll
    for (int s = 0; s < 4; ++s) {
      v16bf bf = load_b_frag(Bp + (size_t)s * 16 * T_SEQ, T_SEQ, k0);
      acc[s] = wmma_bf16(a, bf, acc[s]);
    }
  }
  int lane = threadIdx.x & 31;
  int nn = lane & 15, moff = (lane >> 4) << 3;
  __bf16* yp = y + ((size_t)b * T_SEQ + m0) * C_DIM + n0;
#pragma unroll
  for (int s = 0; s < 4; ++s)
#pragma unroll
    for (int r = 0; r < 8; ++r)
      yp[(size_t)(moff + r) * C_DIM + s * 16 + nn] = (__bf16)acc[s][r];
}

// ======================= K7c: out = y @ W_O^T (TDM + WMMA) ===================
__global__ void __launch_bounds__(256) out_gemm_kernel(
    const __bf16* __restrict__ y, const __bf16* __restrict__ Wo,
    float* __restrict__ out) {
  __shared__ __bf16 Btile[64 * 256];            // 32KB, LDS offset 0
  int wave = threadIdx.x >> 5;
  int m0 = blockIdx.y * 128 + wave * 16;
  int n0 = blockIdx.x * 64;
  stage_b_tile(Btile, Wo + (size_t)n0 * C_DIM);

  const __bf16* Ap = y + (size_t)m0 * C_DIM;
  v8f acc[4] = {};
#pragma unroll
  for (int k0 = 0; k0 < C_DIM; k0 += 32) {
    v16bf a = load_a_frag(Ap, C_DIM, k0);
#pragma unroll
    for (int s = 0; s < 4; ++s) {
      v16bf b = load_b_frag(Btile + s * 16 * 256, 256, k0);
      acc[s] = wmma_bf16(a, b, acc[s]);
    }
  }
  float* base = out + (size_t)m0 * C_DIM + n0;
#pragma unroll
  for (int s = 0; s < 4; ++s) store_tile_f32(base + s * 16, C_DIM, acc[s]);
}

// ============================== launch =======================================
static inline size_t align256(size_t v) { return (v + 255) & ~(size_t)255; }

extern "C" void kernel_launch(void* const* d_in, const int* in_sizes, int n_in,
                              void* d_out, int out_size, void* d_ws,
                              size_t ws_size, hipStream_t stream) {
  const float* x   = (const float*)d_in[0];
  const float* W_Q = (const float*)d_in[1];
  const float* W_V = (const float*)d_in[2];
  const float* W_O = (const float*)d_in[3];
  float* out = (float*)d_out;

  const size_t BTC = (size_t)B_BATCH * T_SEQ * C_DIM;
  char* w = (char*)d_ws;
  size_t off = 0;
  auto alloc = [&](size_t bytes) -> void* {
    void* p = w + off;
    off = align256(off + bytes);
    return p;
  };
  __bf16* xbf   = (__bf16*)alloc(BTC * 2);
  float*  cosT  = (float*)alloc((size_t)T_SEQ * HALF_C * 4);
  float*  sinT  = (float*)alloc((size_t)T_SEQ * HALF_C * 4);
  float*  Ainv  = (float*)alloc(65536 * 4);
  __bf16* Wqpk  = (__bf16*)alloc(65536 * 2);
  __bf16* Wkpk  = (__bf16*)alloc(65536 * 2);
  __bf16* Wvpk  = (__bf16*)alloc(65536 * 2);
  __bf16* Wopk  = (__bf16*)alloc(65536 * 2);
  float*  qraw  = (float*)alloc(BTC * 4);
  float*  kraw  = (float*)alloc(BTC * 4);
  __bf16* qrow  = (__bf16*)alloc(BTC * 2);
  __bf16* krow  = (__bf16*)alloc(BTC * 2);
  __bf16* vT    = (__bf16*)alloc(BTC * 2);
  float*  qcoef = (float*)alloc((size_t)B_BATCH * 7 * C_DIM * 4);
  float*  kcoef = (float*)alloc((size_t)B_BATCH * 7 * C_DIM * 4);
  float*  coef  = (float*)alloc((size_t)B_BATCH * 49 * 4);
  __bf16* P     = (__bf16*)alloc((size_t)B_BATCH * T_SEQ * T_SEQ * 2);
  __bf16* ybf   = (__bf16*)alloc(BTC * 2);
  (void)ws_size; (void)in_sizes; (void)n_in; (void)out_size;

  prep_kernel<<<512, 256, 0, stream>>>(x, xbf, cosT, sinT);
  gj_inverse_kernel<<<1, 256, 0, stream>>>(W_Q, Ainv);
  pack_weights_kernel<<<256, 256, 0, stream>>>(W_Q, Ainv, W_V, W_O,
                                               Wqpk, Wkpk, Wvpk, Wopk);
  gemm_qkv_kernel<<<dim3(4, 64, 3), 256, 0, stream>>>(xbf, Wqpk, Wkpk, Wvpk,
                                                      qraw, kraw, vT);
  rope_norm_kernel<<<B_BATCH * T_SEQ, 256, 0, stream>>>(qraw, kraw, cosT, sinT,
                                                        qrow, krow);
  haar_coef_kernel<<<dim3(7, B_BATCH), 256, 0, stream>>>(qrow, krow, qcoef,
                                                         kcoef);
  coef_gemm_kernel<<<B_BATCH, 256, 0, stream>>>(qcoef, kcoef, coef);
  score_softmax_kernel<<<dim3(T_SEQ / 16, B_BATCH), 256, 0, stream>>>(
      qrow, krow, coef, P);
  attn_v_kernel<<<dim3(C_DIM / 64, T_SEQ / 128, B_BATCH), 256, 0, stream>>>(
      P, vT, ybf);
  out_gemm_kernel<<<dim3(C_DIM / 64, (B_BATCH * T_SEQ) / 128), 256, 0,
                    stream>>>(ybf, Wopk, out);
}
